// DiscoverODEVariableParameters_74148315398270
// MI455X (gfx1250) — matrix operand order repaired and checked by reading, not compiled, not run on hardware
//
#include <hip/hip_runtime.h>

#define DD 128
#define NPAR 16
#define HH 256
#define NCOEF 256
#define BATCH 4096
#define XCOLS 144   // D + NPARAMS

typedef __attribute__((ext_vector_type(16))) _Float16 v16h;
typedef __attribute__((ext_vector_type(8)))  _Float16 v8h;
typedef __attribute__((ext_vector_type(8)))  float    v8f;

// ---------------------------------------------------------------------------
// A-matrix 16x32 f16 fragment (ISA 7.12.2 "16-bit A-Matrix 16x32"):
//   lane l<16 : row M = l,      K = {k0..k0+7, k0+16..k0+23}
//   lane l>=16: row M = l-16,   K = {k0+8..k0+15, k0+24..k0+31}
// ---------------------------------------------------------------------------
__device__ __forceinline__ v16h load_fragA(const _Float16* __restrict__ m,
                                           int ld, int row, int k0, int lane) {
  const _Float16* p = m + (size_t)row * ld + k0 + ((lane & 16) ? 8 : 0);
  v8h lo = *(const v8h*)(p);
  v8h hi = *(const v8h*)(p + 16);
  v16h f;
#pragma unroll
  for (int i = 0; i < 8; ++i) { f[i] = lo[i]; f[i + 8] = hi[i]; }
  return f;
}

// ---------------------------------------------------------------------------
// B-matrix 32x16 f16 fragment, same K order as the A fragment:
//   lane l<16 : col N = l,     K = {k0..k0+7, k0+16..k0+23}
//   lane l>=16: col N = l-16,  K = {k0+8..k0+15, k0+24..k0+31}
// We read column n of B == row n of W (W is N x K row-major, since ref does
// h @ W.T), so each lane loads two 8-half runs of W[n].
// ---------------------------------------------------------------------------
__device__ __forceinline__ v16h load_fragB(const _Float16* __restrict__ m,
                                           int ld, int row, int k0, int lane) {
  const _Float16* p = m + (size_t)row * ld + k0 + ((lane & 16) ? 8 : 0);
  v8h lo = *(const v8h*)(p);
  v8h hi = *(const v8h*)(p + 16);
  v16h f;
#pragma unroll
  for (int i = 0; i < 8; ++i) { f[i] = lo[i]; f[i + 8] = hi[i]; }
  return f;
}

// ---------------------------------------------------------------------------
// C[M,N] = act(A[M,K] @ W[N,K]^T + bias).
// One wave -> 32(M) x 64(N) tile (2 M-subtiles x 4 N-subtiles, 8 accumulators,
// each B fragment reused for both M-subtiles). Block = 128 thr = 4 waves = N=256.
// ---------------------------------------------------------------------------
__global__ void gemm_relu_wmma(const _Float16* __restrict__ A,
                               const _Float16* __restrict__ W,
                               const float* __restrict__ bias,
                               _Float16* __restrict__ outH,  // f16 out (or null)
                               float* __restrict__ outF,     // f32 out (or null)
                               int M, int N, int K, int do_relu) {
  const int lane = threadIdx.x & 31;
  const int wave = threadIdx.x >> 5;
  const int m0 = blockIdx.x * 32;
  const int n0 = wave * 64;
  if (n0 >= N) return;

  v8f acc[2][4] = {};
  const int arow = m0 + (lane & 15);

  for (int k0 = 0; k0 < K; k0 += 32) {
    v16h a0 = load_fragA(A, K, arow, k0, lane);
    v16h a1 = load_fragA(A, K, arow + 16, k0, lane);
#pragma unroll
    for (int t = 0; t < 4; ++t) {
      v16h b = load_fragB(W, K, n0 + t * 16 + (lane & 15), k0, lane);
      acc[0][t] = __builtin_amdgcn_wmma_f32_16x16x32_f16(
          false, a0, false, b, (short)0, acc[0][t], false, false);
      acc[1][t] = __builtin_amdgcn_wmma_f32_16x16x32_f16(
          false, a1, false, b, (short)0, acc[1][t], false, false);
    }
  }

  const int nl = lane & 15;
  const int moff = (lane & 16) ? 8 : 0;
#pragma unroll
  for (int mi = 0; mi < 2; ++mi) {
#pragma unroll
    for (int t = 0; t < 4; ++t) {
      const int n = n0 + t * 16 + nl;
      const float bv = bias[n];
#pragma unroll
      for (int r = 0; r < 8; ++r) {
        float v = acc[mi][t][r] + bv;
        if (do_relu) v = fmaxf(v, 0.0f);
        const size_t idx = (size_t)(m0 + mi * 16 + moff + r) * N + n;
        if (outH) outH[idx] = (_Float16)v;
        if (outF) outF[idx] = v;
      }
    }
  }
}

// ---------------------------------------------------------------------------
// Pack the 16 param columns of x (4096 x 144 f32) into f16, zero-padded to K=32
// ---------------------------------------------------------------------------
__global__ void pack_params(const float* __restrict__ x, _Float16* __restrict__ a16) {
  const int i = blockIdx.x;
  const int k = threadIdx.x;  // 0..31
  const float v = (k < NPAR) ? x[(size_t)i * XCOLS + DD + k] : 0.0f;
  a16[(size_t)i * 32 + k] = (_Float16)v;
}

// f32 (N x K) -> f16 (N x Kp), zero pad K..Kp
__global__ void pack_w(const float* __restrict__ w, _Float16* __restrict__ w16,
                       int N, int K, int Kp) {
  const int idx = blockIdx.x * blockDim.x + threadIdx.x;
  if (idx >= N * Kp) return;
  const int n = idx / Kp, k = idx % Kp;
  w16[idx] = (_Float16)((k < K) ? w[(size_t)n * K + k] : 0.0f);
}

// ---------------------------------------------------------------------------
// RK4 integration of the pendulum ring. One wave32 per batch row, 4 oscillators
// per lane; neighbor coupling via intra-wave shuffles (no LDS, no barriers).
// ---------------------------------------------------------------------------
__device__ __forceinline__ void ring_rhs(const float th[4], const float v[4],
                                         const float o2[4], const float c[4],
                                         const float cr[4], int lane,
                                         float dth[4], float dv[4]) {
  const float left  = __shfl(th[3], (lane + 31) & 31, 32);
  const float right = __shfl(th[0], (lane + 1) & 31, 32);
  const float nl[4] = { left,  th[0], th[1], th[2] };
  const float nr[4] = { th[1], th[2], th[3], right };
#pragma unroll
  for (int j = 0; j < 4; ++j) {
    dth[j] = v[j];
    dv[j] = -__sinf(th[j]) * o2[j] + cr[j] * (nl[j] - th[j]) + c[j] * (nr[j] - th[j]);
  }
}

__global__ void ode_rk4(const float* __restrict__ x,
                        const float* __restrict__ coef,
                        float* __restrict__ out) {
  const int lane = threadIdx.x & 31;
  const int row = blockIdx.x * (blockDim.x >> 5) + (threadIdx.x >> 5);
  const int jb = lane * 4;

  float th[4], vv[4], o2[4], c[4], cr[4];
  const float TWO_PI = 6.28318530717958647692f;
  const float PI = 3.14159265358979323846f;
#pragma unroll
  for (int j = 0; j < 4; ++j) {
    const int g = jb + j;
    th[j] = x[(size_t)row * XCOLS + g] * TWO_PI - PI;     // inputs_phys * ENC_AMP
    vv[j] = 0.0f;
    const float o = coef[(size_t)row * NCOEF + g] * 1.5f + 0.5f;  // W0 range
    o2[j] = o * o;
    c[j] = coef[(size_t)row * NCOEF + DD + g];            // C range is [0,1]
    int fl = row * DD + g - 1;                            // global flat roll by 1
    if (fl < 0) fl += BATCH * DD;
    cr[j] = coef[(size_t)(fl >> 7) * NCOEF + DD + (fl & 127)];
  }

  const int NSUB = 16;
  const int NSTEPS = 59 * NSUB;                 // TS[-1] = 59/30
  const float dt = (1.0f / 30.0f) / (float)NSUB;

  for (int s = 0; s < NSTEPS; ++s) {
    float k1t[4], k1v[4], k2t[4], k2v[4], k3t[4], k3v[4], k4t[4], k4v[4];
    float tt[4], tv[4];
    ring_rhs(th, vv, o2, c, cr, lane, k1t, k1v);
#pragma unroll
    for (int j = 0; j < 4; ++j) { tt[j] = th[j] + 0.5f * dt * k1t[j]; tv[j] = vv[j] + 0.5f * dt * k1v[j]; }
    ring_rhs(tt, tv, o2, c, cr, lane, k2t, k2v);
#pragma unroll
    for (int j = 0; j < 4; ++j) { tt[j] = th[j] + 0.5f * dt * k2t[j]; tv[j] = vv[j] + 0.5f * dt * k2v[j]; }
    ring_rhs(tt, tv, o2, c, cr, lane, k3t, k3v);
#pragma unroll
    for (int j = 0; j < 4; ++j) { tt[j] = th[j] + dt * k3t[j]; tv[j] = vv[j] + dt * k3v[j]; }
    ring_rhs(tt, tv, o2, c, cr, lane, k4t, k4v);
#pragma unroll
    for (int j = 0; j < 4; ++j) {
      th[j] += (dt / 6.0f) * (k1t[j] + 2.0f * k2t[j] + 2.0f * k3t[j] + k4t[j]);
      vv[j] += (dt / 6.0f) * (k1v[j] + 2.0f * k2v[j] + 2.0f * k3v[j] + k4v[j]);
    }
  }

#pragma unroll
  for (int j = 0; j < 4; ++j)
    out[(size_t)row * DD + jb + j] = th[j] * (1.0f / 2.5f);  // (out - B)/A
}

// ---------------------------------------------------------------------------
extern "C" void kernel_launch(void* const* d_in, const int* in_sizes, int n_in,
                              void* d_out, int out_size, void* d_ws, size_t ws_size,
                              hipStream_t stream) {
  const float* x     = (const float*)d_in[0];
  const float* w_in  = (const float*)d_in[1];
  const float* b_in  = (const float*)d_in[2];
  const float* w0    = (const float*)d_in[3];
  const float* b0    = (const float*)d_in[4];
  const float* w1    = (const float*)d_in[5];
  const float* b1    = (const float*)d_in[6];
  const float* w_out = (const float*)d_in[7];
  const float* b_out = (const float*)d_in[8];

  char* ws = (char*)d_ws;
  size_t off = 0;
  _Float16* a16  = (_Float16*)(ws + off); off += (size_t)BATCH * 32 * 2;     // params, K-padded
  _Float16* wi16 = (_Float16*)(ws + off); off += (size_t)HH * 32 * 2;
  _Float16* w016 = (_Float16*)(ws + off); off += (size_t)HH * HH * 2;
  _Float16* w116 = (_Float16*)(ws + off); off += (size_t)HH * HH * 2;
  _Float16* wo16 = (_Float16*)(ws + off); off += (size_t)NCOEF * HH * 2;
  _Float16* h1   = (_Float16*)(ws + off); off += (size_t)BATCH * HH * 2;
  _Float16* h2   = (_Float16*)(ws + off); off += (size_t)BATCH * HH * 2;
  float*    coef = (float*)   (ws + off); off += (size_t)BATCH * NCOEF * 4;

  // Pack inputs/weights to f16
  pack_params<<<BATCH, 32, 0, stream>>>(x, a16);
  pack_w<<<(HH * 32 + 255) / 256, 256, 0, stream>>>(w_in, wi16, HH, NPAR, 32);
  pack_w<<<(HH * HH + 255) / 256, 256, 0, stream>>>(w0, w016, HH, HH, HH);
  pack_w<<<(HH * HH + 255) / 256, 256, 0, stream>>>(w1, w116, HH, HH, HH);
  pack_w<<<(NCOEF * HH + 255) / 256, 256, 0, stream>>>(w_out, wo16, NCOEF, HH, HH);

  // MLP: 4 WMMA GEMMs (bias + ReLU fused; last layer f32 out, no ReLU)
  gemm_relu_wmma<<<BATCH / 32, 128, 0, stream>>>(a16, wi16, b_in, h1, nullptr, BATCH, HH, 32, 1);
  gemm_relu_wmma<<<BATCH / 32, 128, 0, stream>>>(h1, w016, b0, h2, nullptr, BATCH, HH, HH, 1);
  gemm_relu_wmma<<<BATCH / 32, 128, 0, stream>>>(h2, w116, b1, h1, nullptr, BATCH, HH, HH, 1);
  gemm_relu_wmma<<<BATCH / 32, 128, 0, stream>>>(h1, wo16, b_out, nullptr, coef, BATCH, NCOEF, HH, 0);

  // ODE: one wave32 per batch row, 8 rows per 256-thread block
  ode_rk4<<<BATCH / 8, 256, 0, stream>>>(x, coef, (float*)d_out);
}
